// Attention_5712306504144
// MI455X (gfx1250) — matrix operand order repaired and checked by reading, not compiled, not run on hardware
//
#include <hip/hip_runtime.h>
#include <stdint.h>

#define SEQ   2048
#define DIM   2048
#define NH    32
#define NKV   8
#define HD    64
#define KVD   (NKV*HD)   // 512

typedef __attribute__((ext_vector_type(16))) __bf16 v16bf;
typedef __attribute__((ext_vector_type(8)))  float  v8f;

union Frag {
  v16bf v;
  uint4 q[2];
  unsigned short us[16];
};

__device__ __forceinline__ unsigned short f2bf(float f) {
  unsigned int u = __float_as_uint(f);
  u += 0x7FFFu + ((u >> 16) & 1u);   // round-to-nearest-even
  return (unsigned short)(u >> 16);
}

// Async global -> LDS copy (16B per lane), tracked by ASYNCcnt.
__device__ __forceinline__ void async_ld(unsigned ldsoff, unsigned goff,
                                         const void* base) {
  asm volatile("global_load_async_to_lds_b128 %0, %1, %2"
               :: "v"(ldsoff), "v"(goff), "s"(base)
               : "memory");
}
#define ASYNC_WAIT(n) asm volatile("s_wait_asynccnt " #n ::: "memory")

__device__ __forceinline__ unsigned lds_off(const void* p) {
  return (unsigned)(size_t)p;   // shared-aperture low 32 bits = LDS offset
}

// ---------------- elementwise fp32 -> bf16 cast ----------------
__global__ void cast_bf16_kernel(const float* __restrict__ src,
                                 unsigned short* __restrict__ dst, int n) {
  int i = blockIdx.x * blockDim.x + threadIdx.x;
  if (i < n) dst[i] = f2bf(src[i]);
}

// ---------------- rotary embedding (fp32 in, bf16 out) ----------------
__global__ void rope_kernel(const float* __restrict__ in,
                            const float* __restrict__ ct,
                            const float* __restrict__ st,
                            unsigned short* __restrict__ out, int nheads) {
  int idx = blockIdx.x * blockDim.x + threadIdx.x;   // over SEQ*nheads*32
  int tot = SEQ * nheads * (HD / 2);
  if (idx >= tot) return;
  int i = idx % (HD / 2);
  int t = idx / (HD / 2);
  int h = t % nheads;
  int s = t / nheads;
  int base = (s * nheads + h) * HD + 2 * i;
  float r  = in[base];
  float im = in[base + 1];
  float c  = ct[s * (HD / 2) + i];
  float sn = st[s * (HD / 2) + i];
  out[base]     = f2bf(r * c  - im * sn);
  out[base + 1] = f2bf(r * sn + im * c);
}

// ---------------- C(MxN) f32 = A(MxK)bf16 @ B(NxK)bf16^T ----------------
// 256 threads = 8 waves; output tile 128x128; k-step 64, double-buffered
// async-to-LDS staging of A/B slices. Inner 16-step WMMA loop uses a 2-deep
// rotating B-fragment buffer so each WMMA overlaps the next step's DS loads.
__global__ __launch_bounds__(256)
void gemm_nt_kernel(const unsigned short* __restrict__ A,
                    const unsigned short* __restrict__ B,
                    float* __restrict__ C, int M, int N, int K) {
  __shared__ unsigned short sA[2][128 * 64];
  __shared__ unsigned short sB[2][128 * 64];

  int tid  = threadIdx.x;
  int wave = tid >> 5;
  int lane = tid & 31;
  int half = lane >> 4;
  int l16  = lane & 15;
  int rowb = blockIdx.y * 128;
  int colb = blockIdx.x * 128;

  // cooperative fill position: thread -> (row frow, 32 contiguous k at fcol)
  int frow = tid >> 1;
  int fcol = (tid & 1) * 32;

  v8f zero = {};
  v8f acc[8];
#pragma unroll
  for (int t = 0; t < 8; ++t) acc[t] = zero;

  // prologue: fill buffer 0 with k-slice [0,64)
  {
    unsigned lA = lds_off(&sA[0][frow * 64 + fcol]);
    unsigned lB = lds_off(&sB[0][frow * 64 + fcol]);
    unsigned gA = (unsigned)(((rowb + frow) * K + fcol) * 2);
    unsigned gB = (unsigned)(((colb + frow) * K + fcol) * 2);
#pragma unroll
    for (int c = 0; c < 4; ++c) {
      async_ld(lA + c * 16, gA + c * 16, A);
      async_ld(lB + c * 16, gB + c * 16, B);
    }
  }

  int nk = K / 64;
  for (int i = 0; i < nk; ++i) {
    int cur = i & 1;
    if (i + 1 < nk) {
      int kk = (i + 1) * 64;
      unsigned lA = lds_off(&sA[cur ^ 1][frow * 64 + fcol]);
      unsigned lB = lds_off(&sB[cur ^ 1][frow * 64 + fcol]);
      unsigned gA = (unsigned)(((rowb + frow) * K + fcol + kk) * 2);
      unsigned gB = (unsigned)(((colb + frow) * K + fcol + kk) * 2);
#pragma unroll
      for (int c = 0; c < 4; ++c) {
        async_ld(lA + c * 16, gA + c * 16, A);
        async_ld(lB + c * 16, gB + c * 16, B);
      }
      ASYNC_WAIT(8);   // drain current slice; next slice stays in flight
    } else {
      ASYNC_WAIT(0);
    }
    __syncthreads();

    const unsigned short* tA = sA[cur];
    const unsigned short* tB = sB[cur];

    // preload both A fragments (kd = 0,1)
    Frag a[2];
#pragma unroll
    for (int kd = 0; kd < 2; ++kd) {
      const unsigned short* ar = tA + (wave * 16 + l16) * 64 + kd * 32;
      a[kd].q[0] = *(const uint4*)(ar + 16 * half);
      a[kd].q[1] = *(const uint4*)(ar + 16 * half + 8);
    }

    // 16 WMMA steps (u = kd*8 + t) with 2-deep rotating B-fragment pipeline
    Frag b[2];
    {
      const unsigned short* br = tB + l16 * 64 + 16 * half;   // kd=0, t=0
      b[0].q[0] = *(const uint4*)(br);
      b[0].q[1] = *(const uint4*)(br + 8);
    }
#pragma unroll
    for (int u = 0; u < 16; ++u) {
      int kd = u >> 3, t = u & 7;
      if (u + 1 < 16) {
        int kd2 = (u + 1) >> 3, t2 = (u + 1) & 7;
        const unsigned short* br =
            tB + (t2 * 16 + l16) * 64 + kd2 * 32 + 16 * half;
        b[(u + 1) & 1].q[0] = *(const uint4*)(br);
        b[(u + 1) & 1].q[1] = *(const uint4*)(br + 8);
      }
      acc[t] = __builtin_amdgcn_wmma_f32_16x16x32_bf16(
          false, a[kd].v, false, b[u & 1].v, (short)0, acc[t], false, false);
    }
    __syncthreads();
  }

#pragma unroll
  for (int t = 0; t < 8; ++t)
#pragma unroll
    for (int r = 0; r < 8; ++r)
      C[(size_t)(rowb + wave * 16 + r + 8 * half) * N + colb + t * 16 + l16] =
          acc[t][r];
}

// ---------------- flash attention (bf16 in, bf16 out) ----------------
// grid = (16 q-tiles of 128 rows, 32 heads); block = 256 (8 waves).
// K/V 32x64 tiles double-buffered via async-to-LDS; waves skip compute for
// key blocks fully above their causal limit (wave-uniform branch).
__global__ __launch_bounds__(256)
void flash_attn_kernel(const unsigned short* __restrict__ Q,   // [SEQ][NH*HD]
                       const unsigned short* __restrict__ Kc,  // [SEQ][KVD]
                       const unsigned short* __restrict__ Vc,  // [SEQ][KVD]
                       unsigned short* __restrict__ O) {       // [SEQ][NH*HD]
  __shared__ unsigned short lK[2][32 * 64];
  __shared__ unsigned short lV[2][32 * 64];
  __shared__ unsigned short lP[8][16 * 32];

  int qtile = blockIdx.x;
  int h     = blockIdx.y;
  int kvh   = h >> 2;           // GQA: 4 query heads per KV head
  int wave  = threadIdx.x >> 5;
  int lane  = threadIdx.x & 31;
  int half  = lane >> 4;
  int l16   = lane & 15;

  int qbase = qtile * 128 + wave * 16;

  // cooperative fill position: 256 threads * 8 elems = 32x64 tile
  int fidx = threadIdx.x * 8;
  int fcol = fidx & 63;
  unsigned gkv = (unsigned)(((fidx >> 6) * KVD + kvh * HD + fcol) * 2);

  // Q A-fragments for d in [0,32) and [32,64)
  Frag aq[2];
  {
    const unsigned short* qrow = Q + (size_t)(qbase + l16) * (NH * HD) + h * HD;
#pragma unroll
    for (int kd = 0; kd < 2; ++kd) {
      aq[kd].q[0] = *(const uint4*)(qrow + kd * 32 + 16 * half);
      aq[kd].q[1] = *(const uint4*)(qrow + kd * 32 + 16 * half + 8);
    }
  }

  float m_i[8], l_i[8];
  v8f zero = {};
  v8f oacc[4];
#pragma unroll
  for (int r = 0; r < 8; ++r) { m_i[r] = -__builtin_inff(); l_i[r] = 0.f; }
#pragma unroll
  for (int t = 0; t < 4; ++t) oacc[t] = zero;

  int nkb = (qtile + 1) * 4;    // 32-key blocks up to causal limit of tile

  // prologue: fill buffer 0 with key block 0
  async_ld(lds_off(&lK[0][fidx]), gkv, Kc);
  async_ld(lds_off(&lV[0][fidx]), gkv, Vc);

  for (int kb = 0; kb < nkb; ++kb) {
    int cur   = kb & 1;
    int kbase = kb * 32;
    if (kb + 1 < nkb) {
      unsigned go = gkv + (unsigned)((kb + 1) * 32 * KVD * 2);
      async_ld(lds_off(&lK[cur ^ 1][fidx]), go, Kc);
      async_ld(lds_off(&lV[cur ^ 1][fidx]), go, Vc);
      ASYNC_WAIT(2);
    } else {
      ASYNC_WAIT(0);
    }
    __syncthreads();

    if (kbase <= qbase + 15) {  // wave-uniform causal skip
      // preload all four K fragments (2 key subtiles x 2 k-chunks)
      Frag bk0[2], bk1[2];
#pragma unroll
      for (int t = 0; t < 2; ++t) {
        const unsigned short* kr = &lK[cur][(t * 16 + l16) * 64 + 16 * half];
        bk0[t].q[0] = *(const uint4*)(kr);
        bk0[t].q[1] = *(const uint4*)(kr + 8);
        bk1[t].q[0] = *(const uint4*)(kr + 32);
        bk1[t].q[1] = *(const uint4*)(kr + 40);
      }
      // scores: 16 rows x 32 keys = two 16x16 tiles, k-dim = HD = 2x32
      v8f sc[2];
      sc[0] = __builtin_amdgcn_wmma_f32_16x16x32_bf16(
          false, aq[0].v, false, bk0[0].v, (short)0, zero, false, false);
      sc[1] = __builtin_amdgcn_wmma_f32_16x16x32_bf16(
          false, aq[0].v, false, bk0[1].v, (short)0, zero, false, false);
      sc[0] = __builtin_amdgcn_wmma_f32_16x16x32_bf16(
          false, aq[1].v, false, bk1[0].v, (short)0, sc[0], false, false);
      sc[1] = __builtin_amdgcn_wmma_f32_16x16x32_bf16(
          false, aq[1].v, false, bk1[1].v, (short)0, sc[1], false, false);

      // causal mask + online softmax (rows in r+8*half, cols in lane%16)
      float p0v[8], p1v[8];
#pragma unroll
      for (int r = 0; r < 8; ++r) {
        int qg = qbase + r + 8 * half;
        float v0 = sc[0][r] * 0.125f;             // 1/sqrt(64)
        float v1 = sc[1][r] * 0.125f;
        bool msk0 = (kbase + l16) > qg;
        bool msk1 = (kbase + 16 + l16) > qg;
        if (msk0) v0 = -__builtin_inff();
        if (msk1) v1 = -__builtin_inff();
        float mx = fmaxf(v0, v1);
#pragma unroll
        for (int off = 1; off < 16; off <<= 1)
          mx = fmaxf(mx, __shfl_xor(mx, off, 32));
        float mn = fmaxf(m_i[r], mx);
        float p0 = msk0 ? 0.f : __expf(v0 - mn);
        float p1 = msk1 ? 0.f : __expf(v1 - mn);
        p0v[r] = p0; p1v[r] = p1;
        float rs = p0 + p1;
#pragma unroll
        for (int off = 1; off < 16; off <<= 1)
          rs += __shfl_xor(rs, off, 32);
        float alpha = (mn == m_i[r]) ? 1.f : __expf(m_i[r] - mn);
        l_i[r] = l_i[r] * alpha + rs;
        m_i[r] = mn;
#pragma unroll
        for (int t = 0; t < 4; ++t) oacc[t][r] *= alpha;
      }
      // round-trip P through per-wave LDS: C-layout -> A-layout
#pragma unroll
      for (int r = 0; r < 8; ++r) {
        lP[wave][(r + 8 * half) * 32 + l16]      = f2bf(p0v[r]);
        lP[wave][(r + 8 * half) * 32 + 16 + l16] = f2bf(p1v[r]);
      }
      asm volatile("s_wait_dscnt 0x0" ::: "memory");
      Frag ap;
      {
        const unsigned short* pr = &lP[wave][l16 * 32 + 16 * half];
        ap.q[0] = *(const uint4*)(pr);
        ap.q[1] = *(const uint4*)(pr + 8);
      }
      // P @ V : 2-deep rotating V-fragment pipeline over 4 output tiles
      Frag bv[2];
#pragma unroll
      for (int e = 0; e < 16; ++e)
        bv[0].us[e] = lV[cur][(e + 16 * half) * 64 + l16];
#pragma unroll
      for (int t = 0; t < 4; ++t) {
        if (t + 1 < 4) {
#pragma unroll
          for (int e = 0; e < 16; ++e)
            bv[(t + 1) & 1].us[e] =
                lV[cur][(e + 16 * half) * 64 + (t + 1) * 16 + l16];
        }
        oacc[t] = __builtin_amdgcn_wmma_f32_16x16x32_bf16(
            false, ap.v, false, bv[t & 1].v, (short)0, oacc[t], false, false);
      }
    }
    __syncthreads();
  }

  // epilogue: O / l, write bf16
#pragma unroll
  for (int t = 0; t < 4; ++t)
#pragma unroll
    for (int r = 0; r < 8; ++r) {
      float li  = l_i[r];
      float val = (li > 0.f) ? oacc[t][r] / li : 0.f;
      O[(size_t)(qbase + r + 8 * half) * (NH * HD) + h * HD + t * 16 + l16] =
          f2bf(val);
    }
}

// ---------------- host orchestration ----------------
extern "C" void kernel_launch(void* const* d_in, const int* in_sizes, int n_in,
                              void* d_out, int out_size, void* d_ws, size_t ws_size,
                              hipStream_t stream) {
  const float* x  = (const float*)d_in[0];
  const float* fc = (const float*)d_in[1];
  const float* fs = (const float*)d_in[2];
  const float* wq = (const float*)d_in[3];
  const float* wk = (const float*)d_in[4];
  const float* wv = (const float*)d_in[5];
  const float* wo = (const float*)d_in[6];
  float* out = (float*)d_out;

  char* ws = (char*)d_ws;
  size_t off = 0;
  auto alloc = [&](size_t bytes) -> void* {
    void* p = ws + off;
    off += (bytes + 255) & ~(size_t)255;
    return p;
  };

  unsigned short* xb  = (unsigned short*)alloc((size_t)SEQ * DIM * 2);
  unsigned short* wqb = (unsigned short*)alloc((size_t)DIM * DIM * 2);
  unsigned short* wkb = (unsigned short*)alloc((size_t)KVD * DIM * 2);
  unsigned short* wvb = (unsigned short*)alloc((size_t)KVD * DIM * 2);
  unsigned short* wob = (unsigned short*)alloc((size_t)DIM * DIM * 2);
  float*          qf  = (float*)alloc((size_t)SEQ * DIM * 4);
  float*          kf  = (float*)alloc((size_t)SEQ * KVD * 4);
  float*          vf  = (float*)alloc((size_t)SEQ * KVD * 4);
  unsigned short* qb  = (unsigned short*)alloc((size_t)SEQ * DIM * 2);
  unsigned short* kb  = (unsigned short*)alloc((size_t)SEQ * KVD * 2);
  unsigned short* vb  = (unsigned short*)alloc((size_t)SEQ * KVD * 2);
  unsigned short* ab  = (unsigned short*)alloc((size_t)SEQ * DIM * 2);

  // 1) casts to bf16
  cast_bf16_kernel<<<(SEQ * DIM) / 256, 256, 0, stream>>>(x, xb, SEQ * DIM);
  cast_bf16_kernel<<<(DIM * DIM) / 256, 256, 0, stream>>>(wq, wqb, DIM * DIM);
  cast_bf16_kernel<<<(KVD * DIM) / 256, 256, 0, stream>>>(wk, wkb, KVD * DIM);
  cast_bf16_kernel<<<(KVD * DIM) / 256, 256, 0, stream>>>(wv, wvb, KVD * DIM);
  cast_bf16_kernel<<<(DIM * DIM) / 256, 256, 0, stream>>>(wo, wob, DIM * DIM);

  // 2) projections: Q = X Wq^T, K = X Wk^T, V = X Wv^T
  gemm_nt_kernel<<<dim3(DIM / 128, SEQ / 128), 256, 0, stream>>>(xb, wqb, qf, SEQ, DIM, DIM);
  gemm_nt_kernel<<<dim3(KVD / 128, SEQ / 128), 256, 0, stream>>>(xb, wkb, kf, SEQ, KVD, DIM);
  gemm_nt_kernel<<<dim3(KVD / 128, SEQ / 128), 256, 0, stream>>>(xb, wvb, vf, SEQ, KVD, DIM);

  // 3) rotary (Q, K) + cast (V) to bf16
  rope_kernel<<<(SEQ * NH * (HD / 2)) / 256, 256, 0, stream>>>(qf, fc, fs, qb, NH);
  rope_kernel<<<(SEQ * NKV * (HD / 2)) / 256, 256, 0, stream>>>(kf, fc, fs, kb, NKV);
  cast_bf16_kernel<<<(SEQ * KVD) / 256, 256, 0, stream>>>(vf, vb, SEQ * KVD);

  // 4) causal flash attention
  flash_attn_kernel<<<dim3(SEQ / 128, NH), 256, 0, stream>>>(qb, kb, vb, ab);

  // 5) output projection: out = attn Wo^T (fp32 result)
  gemm_nt_kernel<<<dim3(DIM / 128, SEQ / 128), 256, 0, stream>>>(ab, wob, out, SEQ, DIM, DIM);
}
